// WorldStateRepresentation_6236292514553
// MI455X (gfx1250) — matrix operand order repaired and checked by reading, not compiled, not run on hardware
//
#include <hip/hip_runtime.h>
#include <hip/hip_bf16.h>

#define DEVINL __device__ __forceinline__

typedef __bf16 bf16_t;
typedef __attribute__((ext_vector_type(16))) __bf16 v16bf;
typedef __attribute__((ext_vector_type(8)))  float  v8f;

static constexpr int kD   = 512;    // world state dim
static constexpr int kM   = 1000;   // memory rows
static constexpr int kMP  = 1024;   // memory rows padded (ctx GEMM K)
static constexpr int kMT  = 63;     // ceil(1000/16) row tiles
static constexpr int kB   = 8192;   // batch
static constexpr int kXD  = 544;    // 528 (D+16) padded to mult of 32
static constexpr int kXDr = 528;

// ---------------- WMMA fragment helpers (bf16 16x16x32, wave32) ----------------
// A (16x32): lane L<16 -> row L, K 0..15; lane>=16 -> row L-16, K 16..31
DEVINL v16bf afrag_bf16(const bf16_t* rowbase, int lane) {
  const int off = (lane >> 4) << 4;
  v16bf a;
#pragma unroll
  for (int i = 0; i < 16; ++i) a[i] = rowbase[off + i];
  return a;
}
// B (32x16): lane L<16 -> col L, K 0..15; lane>=16 -> col L-16, K 16..31
DEVINL v16bf bfrag_bf16(const bf16_t* rowbase, int lane) {
  const int off = (lane >> 4) << 4;
  v16bf b;
#pragma unroll
  for (int i = 0; i < 16; ++i) b[i] = rowbase[off + i];
  return b;
}
DEVINL v8f wmma_bf16(v16bf a, v16bf b, v8f c) {
  return __builtin_amdgcn_wmma_f32_16x16x32_bf16(false, a, false, b, (short)0, c,
                                                 false, false);
}

// Cooperative fp32 -> bf16 staging of a contiguous block into LDS (256 threads)
DEVINL void stage_bf16(const float* __restrict__ src, bf16_t* __restrict__ dst,
                       int nelem, int t) {
  for (int i = t; i < nelem / 4; i += 256) {
    const float4 v = ((const float4*)src)[i];
    const int o = i * 4;
    dst[o]     = (bf16_t)v.x;
    dst[o + 1] = (bf16_t)v.y;
    dst[o + 2] = (bf16_t)v.z;
    dst[o + 3] = (bf16_t)v.w;
  }
}

// Block-wide LayerNorm over LDS hF[16][512] -> hB bf16. 256 threads.
DEVINL void block_layernorm(float* hF, bf16_t* hB, const float* g, const float* beta,
                            float* redS, float* redQ, float* mu, float* rv, int t) {
  const int row = t >> 4, seg = t & 15;
  float s = 0.f, s2 = 0.f;
#pragma unroll 4
  for (int c0 = seg * 32; c0 < seg * 32 + 32; ++c0) {
    float v = hF[row * kD + c0]; s += v; s2 += v * v;
  }
  redS[row * 16 + seg] = s; redQ[row * 16 + seg] = s2;
  __syncthreads();
  if (seg == 0) {
    float S = 0.f, Q = 0.f;
#pragma unroll
    for (int i = 0; i < 16; ++i) { S += redS[row * 16 + i]; Q += redQ[row * 16 + i]; }
    float m = S * (1.f / 512.f);
    float var = Q * (1.f / 512.f) - m * m;
    mu[row] = m; rv[row] = rsqrtf(var + 1e-5f);
  }
  __syncthreads();
  float m = mu[row], r = rv[row];
#pragma unroll 4
  for (int c0 = seg * 32; c0 < seg * 32 + 32; ++c0) {
    float v = (hF[row * kD + c0] - m) * r * g[c0] + beta[c0];
    hB[row * kD + c0] = (bf16_t)v;
  }
  __syncthreads();
}

// ---------------- Kernel 0: one-shot fp32 -> bf16 weight conversion ----------------
__global__ void f32_to_bf16_kernel(const float* __restrict__ src,
                                   bf16_t* __restrict__ dst, int n) {
  int i = (blockIdx.x * blockDim.x + threadIdx.x) * 4;
  if (i + 3 < n) {
    const float4 v = *(const float4*)(src + i);
    dst[i]     = (bf16_t)v.x;
    dst[i + 1] = (bf16_t)v.y;
    dst[i + 2] = (bf16_t)v.z;
    dst[i + 3] = (bf16_t)v.w;
  } else {
    for (; i < n; ++i) dst[i] = (bf16_t)src[i];
  }
}

// ---------------- Kernel 1: K/V projection of memory bank ----------------
// kmat bf16 [1008][512] (rows >=1000 zero); vt bf16 [512][1024]
__global__ void kv_proj_kernel(const float* __restrict__ mem,
                               const bf16_t* __restrict__ wkB, const float* __restrict__ bk,
                               const bf16_t* __restrict__ wvB, const float* __restrict__ bv,
                               bf16_t* __restrict__ kmat, bf16_t* __restrict__ vt) {
  extern __shared__ char smem[];
  bf16_t* aB = (bf16_t*)smem;                       // 16*512*2
  const int t = threadIdx.x, lane = t & 31, wave = t >> 5;
  const int m0 = blockIdx.x * 16;
  // stage 16 (clamped) memory rows as bf16 into LDS
  for (int i = t; i < 16 * kD; i += 256) {
    const int r = i >> 9, c = i & 511;
    const int mr = (m0 + r) < kM ? (m0 + r) : (kM - 1);
    aB[i] = (bf16_t)mem[(size_t)mr * kD + c];
  }
  __syncthreads();

  v8f ck[4] = {}; v8f cv[4] = {};
  for (int k0 = 0; k0 < kD; k0 += 32) {
    v16bf a = afrag_bf16(aB + (lane & 15) * kD + k0, lane);
#pragma unroll
    for (int j = 0; j < 4; ++j) {
      const int n = (wave * 4 + j) * 16 + (lane & 15);
      ck[j] = wmma_bf16(a, bfrag_bf16(wkB + (size_t)n * kD + k0, lane), ck[j]);
      cv[j] = wmma_bf16(a, bfrag_bf16(wvB + (size_t)n * kD + k0, lane), cv[j]);
    }
  }
  const int col = lane & 15, rb = (lane >> 4) * 8;
#pragma unroll
  for (int j = 0; j < 4; ++j) {
    const int n = (wave * 4 + j) * 16 + col;
    const float bkv = bk[n], bvv = bv[n];
#pragma unroll
    for (int i = 0; i < 8; ++i) {
      const int m = m0 + rb + i;
      const bool valid = m < kM;
      kmat[(size_t)m * kD + n] = valid ? (bf16_t)(ck[j][i] + bkv) : (bf16_t)0.0f;
      vt[(size_t)n * kMP + m]  = valid ? (bf16_t)(cv[j][i] + bvv) : (bf16_t)0.0f;
    }
  }
}

// ---------------- Kernel 2: fused encoder (GEMM->ReLU->LN->GEMM) ----------------
__global__ void encoder_kernel(const float* __restrict__ obs,
                               const bf16_t* __restrict__ w1B, const float* __restrict__ b1,
                               const float* __restrict__ g,  const float* __restrict__ beta,
                               const bf16_t* __restrict__ w2B, const float* __restrict__ b2,
                               float* __restrict__ enc) {
  extern __shared__ char smem[];
  bf16_t* aB   = (bf16_t*)smem;                      // 16*512*2 = 16384
  float*  hF   = (float*)(smem + 16384);             // 32768
  bf16_t* hB   = (bf16_t*)(smem + 16384 + 32768);    // 16384
  float*  redS = (float*)(smem + 16384 + 32768 + 16384);
  float*  redQ = redS + 256;
  float*  mu   = redQ + 256;
  float*  rv   = mu + 16;
  const int t = threadIdx.x, lane = t & 31, wave = t >> 5;
  const int b0 = blockIdx.x * 16;
  const int col = lane & 15, rb = (lane >> 4) * 8;

  stage_bf16(obs + (size_t)b0 * kD, aB, 16 * kD, t);
  __syncthreads();

  v8f c[4] = {};
  for (int k0 = 0; k0 < kD; k0 += 32) {
    v16bf a = afrag_bf16(aB + (lane & 15) * kD + k0, lane);
#pragma unroll
    for (int j = 0; j < 4; ++j) {
      const int n = (wave * 4 + j) * 16 + (lane & 15);
      c[j] = wmma_bf16(a, bfrag_bf16(w1B + (size_t)n * kD + k0, lane), c[j]);
    }
  }
#pragma unroll
  for (int j = 0; j < 4; ++j) {
    const int n = (wave * 4 + j) * 16 + col;
    const float bb = b1[n];
#pragma unroll
    for (int i = 0; i < 8; ++i) {
      float v = c[j][i] + bb;
      hF[(rb + i) * kD + n] = v > 0.f ? v : 0.f;
    }
  }
  __syncthreads();
  block_layernorm(hF, hB, g, beta, redS, redQ, mu, rv, t);

  v8f c2[4] = {};
  for (int k0 = 0; k0 < kD; k0 += 32) {
    v16bf a = afrag_bf16(hB + (lane & 15) * kD + k0, lane);
#pragma unroll
    for (int j = 0; j < 4; ++j) {
      const int n = (wave * 4 + j) * 16 + (lane & 15);
      c2[j] = wmma_bf16(a, bfrag_bf16(w2B + (size_t)n * kD + k0, lane), c2[j]);
    }
  }
#pragma unroll
  for (int j = 0; j < 4; ++j) {
    const int n = (wave * 4 + j) * 16 + col;
    const float bb = b2[n];
#pragma unroll
    for (int i = 0; i < 8; ++i)
      enc[(size_t)(b0 + rb + i) * kD + n] = c2[j][i] + bb;
  }
}

// ---------------- Kernel 3: Q projection (scale 1/8 folded in) ----------------
__global__ void qproj_kernel(const float* __restrict__ enc,
                             const bf16_t* __restrict__ wqB, const float* __restrict__ bq,
                             bf16_t* __restrict__ qmat) {
  extern __shared__ char smem[];
  bf16_t* aB = (bf16_t*)smem;                        // 16*512*2
  const int t = threadIdx.x, lane = t & 31, wave = t >> 5;
  const int b0 = blockIdx.x * 16;

  stage_bf16(enc + (size_t)b0 * kD, aB, 16 * kD, t);
  __syncthreads();

  v8f c[4] = {};
  for (int k0 = 0; k0 < kD; k0 += 32) {
    v16bf a = afrag_bf16(aB + (lane & 15) * kD + k0, lane);
#pragma unroll
    for (int j = 0; j < 4; ++j) {
      const int n = (wave * 4 + j) * 16 + (lane & 15);
      c[j] = wmma_bf16(a, bfrag_bf16(wqB + (size_t)n * kD + k0, lane), c[j]);
    }
  }
  const int col = lane & 15, rb = (lane >> 4) * 8;
#pragma unroll
  for (int j = 0; j < 4; ++j) {
    const int n = (wave * 4 + j) * 16 + col;
    const float bb = bq[n];
#pragma unroll
    for (int i = 0; i < 8; ++i)
      qmat[(size_t)(b0 + rb + i) * kD + n] = (bf16_t)((c[j][i] + bb) * 0.125f);
  }
}

// ---------------- Kernel 4: cross-attention (scores -> softmax -> ctx) ----------------
__global__ void attn_kernel(const bf16_t* __restrict__ qmat,
                            const bf16_t* __restrict__ kmat,
                            const bf16_t* __restrict__ vt,
                            bf16_t* __restrict__ ctx) {
  extern __shared__ char smem[];
  float*  S    = (float*)smem;                   // 16*1024*4
  bf16_t* P    = (bf16_t*)(smem + 65536);        // 16*1024*2
  float*  red  = (float*)(smem + 65536 + 32768); // 256
  float*  rst  = red + 256;                      // 16
  const int t = threadIdx.x, lane = t & 31, wave = t >> 5;
  const int b0 = blockIdx.x * 16, h = blockIdx.y;
  const int col = lane & 15, rb = (lane >> 4) * 8;

  // scores: S[r][m] = (q/8) . k
  for (int mt = wave; mt < kMT; mt += 8) {
    v8f c = {};
    const bf16_t* qrow = qmat + (size_t)(b0 + (lane & 15)) * kD + h * 64;
    const bf16_t* krow = kmat + (size_t)(mt * 16 + (lane & 15)) * kD + h * 64;
#pragma unroll
    for (int k0 = 0; k0 < 64; k0 += 32)
      c = wmma_bf16(afrag_bf16(qrow + k0, lane), bfrag_bf16(krow + k0, lane), c);
#pragma unroll
    for (int i = 0; i < 8; ++i)
      S[(rb + i) * kMP + mt * 16 + col] = c[i];
  }
  __syncthreads();

  // two-pass softmax over m<1000
  const int row = t >> 4, seg = t & 15;
  float mx = -1e30f;
  for (int m = seg * 64; m < seg * 64 + 64; ++m)
    if (m < kM) mx = fmaxf(mx, S[row * kMP + m]);
  red[row * 16 + seg] = mx;
  __syncthreads();
  if (seg == 0) {
    float m2 = -1e30f;
#pragma unroll
    for (int i = 0; i < 16; ++i) m2 = fmaxf(m2, red[row * 16 + i]);
    rst[row] = m2;
  }
  __syncthreads();
  float rmax = rst[row], sum = 0.f;
  for (int m = seg * 64; m < seg * 64 + 64; ++m)
    if (m < kM) { float e = __expf(S[row * kMP + m] - rmax); S[row * kMP + m] = e; sum += e; }
  red[row * 16 + seg] = sum;
  __syncthreads();
  if (seg == 0) {
    float tot = 0.f;
#pragma unroll
    for (int i = 0; i < 16; ++i) tot += red[row * 16 + i];
    rst[row] = 1.f / tot;
  }
  __syncthreads();
  float rs = rst[row];
  for (int m = seg * 64; m < seg * 64 + 64; ++m)
    P[row * kMP + m] = (m < kM) ? (bf16_t)(S[row * kMP + m] * rs) : (bf16_t)0.0f;
  __syncthreads();

  // ctx[r][d] = P . V  (waves 0..3, one 16-col tile each of the 64 head dims)
  if (wave < 4) {
    v8f c = {};
    const int n0 = wave * 16;
    for (int k0 = 0; k0 < kMP; k0 += 32) {
      v16bf a = afrag_bf16(P + (lane & 15) * kMP + k0, lane);
      v16bf b = bfrag_bf16(vt + (size_t)(h * 64 + n0 + (lane & 15)) * kMP + k0, lane);
      c = wmma_bf16(a, b, c);
    }
#pragma unroll
    for (int i = 0; i < 8; ++i)
      ctx[(size_t)(b0 + rb + i) * kD + h * 64 + n0 + col] = (bf16_t)c[i];
  }
}

// ---------------- Kernel 5: out-proj + fuse + transition + quality ----------------
__global__ void transition_kernel(const bf16_t* __restrict__ ctx, const float* __restrict__ enc,
                                  const float* __restrict__ ego,
                                  const bf16_t* __restrict__ woB, const float* __restrict__ bo,
                                  const bf16_t* __restrict__ tw1B, const float* __restrict__ tb1,
                                  const float* __restrict__ tg,  const float* __restrict__ tbeta,
                                  const bf16_t* __restrict__ tw2B, const float* __restrict__ tb2,
                                  const bf16_t* __restrict__ qw1B, const float* __restrict__ qb1,
                                  const float* __restrict__ qw2, const float* __restrict__ qb2,
                                  float* __restrict__ out_ns, float* __restrict__ out_q) {
  extern __shared__ char smem[];
  bf16_t* xb  = (bf16_t*)smem;                                      // 16*544*2 = 17408
  float*  hF  = (float*)(smem + 17408);                             // 32768
  bf16_t* hB  = (bf16_t*)(smem + 17408 + 32768);                    // 16384
  bf16_t* nB  = (bf16_t*)(smem + 17408 + 32768 + 16384);            // 16384
  float*  qh  = (float*)(smem + 17408 + 32768 + 16384 + 16384);     // 8192
  float*  redS = (float*)(smem + 17408 + 32768 + 16384 + 16384 + 8192);
  float*  redQ = redS + 256;
  float*  mu   = redQ + 256;
  float*  rv   = mu + 16;
  const int t = threadIdx.x, lane = t & 31, wave = t >> 5;
  const int b0 = blockIdx.x * 16;
  const int col = lane & 15, rb = (lane >> 4) * 8;

  // Stage A: memory_states = ctx @ wo^T + bo ; fused = 0.7*enc + 0.3*ms -> xb
  {
    v8f c[4] = {};
    for (int k0 = 0; k0 < kD; k0 += 32) {
      v16bf a = afrag_bf16(ctx + (size_t)(b0 + (lane & 15)) * kD + k0, lane);
#pragma unroll
      for (int j = 0; j < 4; ++j) {
        const int n = (wave * 4 + j) * 16 + (lane & 15);
        c[j] = wmma_bf16(a, bfrag_bf16(woB + (size_t)n * kD + k0, lane), c[j]);
      }
    }
#pragma unroll
    for (int j = 0; j < 4; ++j) {
      const int n = (wave * 4 + j) * 16 + col;
      const float bb = bo[n];
#pragma unroll
      for (int i = 0; i < 8; ++i) {
        const int r = rb + i;
        float ms = c[j][i] + bb;
        float f = 0.7f * enc[(size_t)(b0 + r) * kD + n] + 0.3f * ms;
        xb[r * kXD + n] = (bf16_t)f;
      }
    }
    const int row = t >> 4, cc = t & 15;
    xb[row * kXD + kD + cc]   = (bf16_t)ego[(size_t)(b0 + row) * 16 + cc];
    xb[row * kXD + kXDr + cc] = (bf16_t)0.0f;   // pad 528..543
  }
  __syncthreads();

  // Stage B: h2 = relu(x @ tr_w1^T + tb1)   (K=544, zero-padded, W guarded)
  {
    v8f c[4] = {};
    for (int k0 = 0; k0 < kXD; k0 += 32) {
      v16bf a = afrag_bf16(xb + (lane & 15) * kXD + k0, lane);
      const int koff = k0 + ((lane >> 4) << 4);
#pragma unroll
      for (int j = 0; j < 4; ++j) {
        const int n = (wave * 4 + j) * 16 + (lane & 15);
        v16bf b = {};
        if (koff < kXDr) b = bfrag_bf16(tw1B + (size_t)n * kXDr + k0, lane);
        c[j] = wmma_bf16(a, b, c[j]);
      }
    }
#pragma unroll
    for (int j = 0; j < 4; ++j) {
      const int n = (wave * 4 + j) * 16 + col;
      const float bb = tb1[n];
#pragma unroll
      for (int i = 0; i < 8; ++i) {
        float v = c[j][i] + bb;
        hF[(rb + i) * kD + n] = v > 0.f ? v : 0.f;
      }
    }
  }
  __syncthreads();
  block_layernorm(hF, hB, tg, tbeta, redS, redQ, mu, rv, t);

  // Stage C: next_state = LN(h2) @ tr_w2^T + tb2
  {
    v8f c[4] = {};
    for (int k0 = 0; k0 < kD; k0 += 32) {
      v16bf a = afrag_bf16(hB + (lane & 15) * kD + k0, lane);
#pragma unroll
      for (int j = 0; j < 4; ++j) {
        const int n = (wave * 4 + j) * 16 + (lane & 15);
        c[j] = wmma_bf16(a, bfrag_bf16(tw2B + (size_t)n * kD + k0, lane), c[j]);
      }
    }
#pragma unroll
    for (int j = 0; j < 4; ++j) {
      const int n = (wave * 4 + j) * 16 + col;
      const float bb = tb2[n];
#pragma unroll
      for (int i = 0; i < 8; ++i) {
        float v = c[j][i] + bb;
        out_ns[(size_t)(b0 + rb + i) * kD + n] = v;
        nB[(rb + i) * kD + n] = (bf16_t)v;
      }
    }
  }
  __syncthreads();

  // Stage D: quality head (512->128 relu, 128->1 sigmoid)
  {
    v8f c = {};
    for (int k0 = 0; k0 < kD; k0 += 32) {
      v16bf a = afrag_bf16(nB + (lane & 15) * kD + k0, lane);
      v16bf b = bfrag_bf16(qw1B + (size_t)(wave * 16 + (lane & 15)) * kD + k0, lane);
      c = wmma_bf16(a, b, c);
    }
    const float bb = qb1[wave * 16 + col];
#pragma unroll
    for (int i = 0; i < 8; ++i)
      qh[(rb + i) * 128 + wave * 16 + col] = fmaxf(c[i] + bb, 0.f);
  }
  __syncthreads();
  if (t < 16) {
    float acc = qb2[0];
#pragma unroll 8
    for (int j = 0; j < 128; ++j) acc += qh[t * 128 + j] * qw2[j];
    out_q[b0 + t] = 1.f / (1.f + __expf(-acc));
  }
}

// ---------------- Kernel 6: sequential scatter-EMA memory update ----------------
__global__ void memupd_kernel(const float* __restrict__ wmem, const float* __restrict__ ns,
                              const float* __restrict__ qual, const int* __restrict__ uidx,
                              float* __restrict__ out_mem) {
  const int row = blockIdx.x;      // 0..999
  const int t = threadIdx.x;       // 0..127, 4 cols each
  float v[4];
#pragma unroll
  for (int c = 0; c < 4; ++c) v[c] = wmem[(size_t)row * kD + t * 4 + c];
  for (int i = 0; i < kB; ++i) {
    const int idx = uidx[i];
    if (idx == row) {
      if (qual[i] > 0.7f) {
#pragma unroll
        for (int c = 0; c < 4; ++c)
          v[c] = 0.9f * v[c] + 0.1f * ns[(size_t)i * kD + t * 4 + c];
      }
    }
  }
#pragma unroll
  for (int c = 0; c < 4; ++c) out_mem[(size_t)row * kD + t * 4 + c] = v[c];
}

// ---------------- host launch ----------------
extern "C" void kernel_launch(void* const* d_in, const int* in_sizes, int n_in,
                              void* d_out, int out_size, void* d_ws, size_t ws_size,
                              hipStream_t stream) {
  (void)in_sizes; (void)n_in; (void)out_size; (void)ws_size;
  const float* obs    = (const float*)d_in[0];
  const float* ego    = (const float*)d_in[1];
  const float* wmem   = (const float*)d_in[2];
  const float* enc_w1 = (const float*)d_in[3];
  const float* enc_b1 = (const float*)d_in[4];
  const float* enc_g  = (const float*)d_in[5];
  const float* enc_be = (const float*)d_in[6];
  const float* enc_w2 = (const float*)d_in[7];
  const float* enc_b2 = (const float*)d_in[8];
  const float* wq = (const float*)d_in[9];  const float* bq = (const float*)d_in[10];
  const float* wk = (const float*)d_in[11]; const float* bk = (const float*)d_in[12];
  const float* wv = (const float*)d_in[13]; const float* bv = (const float*)d_in[14];
  const float* wo = (const float*)d_in[15]; const float* bo = (const float*)d_in[16];
  const float* tw1 = (const float*)d_in[17]; const float* tb1 = (const float*)d_in[18];
  const float* tg  = (const float*)d_in[19]; const float* tbe = (const float*)d_in[20];
  const float* tw2 = (const float*)d_in[21]; const float* tb2 = (const float*)d_in[22];
  const float* qw1 = (const float*)d_in[23]; const float* qb1 = (const float*)d_in[24];
  const float* qw2 = (const float*)d_in[25]; const float* qb2 = (const float*)d_in[26];
  const int*   uidx = (const int*)d_in[27];

  // workspace layout (bytes)
  char* ws = (char*)d_ws;
  size_t off = 0;
  bf16_t* kmat = (bf16_t*)(ws + off); off += (size_t)1008 * kD * 2;   // 1032192
  bf16_t* vt   = (bf16_t*)(ws + off); off += (size_t)kD * kMP * 2;    // 1048576
  float*  enc  = (float*)(ws + off);  off += (size_t)kB * kD * 4;     // 16777216
  bf16_t* qmat = (bf16_t*)(ws + off); off += (size_t)kB * kD * 2;     // 8388608
  bf16_t* ctx  = (bf16_t*)(ws + off); off += (size_t)kB * kD * 2;     // 8388608
  bf16_t* w1B  = (bf16_t*)(ws + off); off += (size_t)kD * kD * 2;
  bf16_t* w2B  = (bf16_t*)(ws + off); off += (size_t)kD * kD * 2;
  bf16_t* wqB  = (bf16_t*)(ws + off); off += (size_t)kD * kD * 2;
  bf16_t* wkB  = (bf16_t*)(ws + off); off += (size_t)kD * kD * 2;
  bf16_t* wvB  = (bf16_t*)(ws + off); off += (size_t)kD * kD * 2;
  bf16_t* woB  = (bf16_t*)(ws + off); off += (size_t)kD * kD * 2;
  bf16_t* tw2B = (bf16_t*)(ws + off); off += (size_t)kD * kD * 2;
  bf16_t* tw1B = (bf16_t*)(ws + off); off += (size_t)kD * kXDr * 2;
  bf16_t* qw1B = (bf16_t*)(ws + off); off += (size_t)128 * kD * 2;

  float* out_ns  = (float*)d_out;
  float* out_q   = out_ns + (size_t)kB * kD;
  float* out_mem = out_q + kB;

  // one-shot weight conversions (all reused by 512 row-blocks; live in L2)
  const int nW = kD * kD;
  const int gW = (nW / 4 + 255) / 256;
  f32_to_bf16_kernel<<<gW, 256, 0, stream>>>(enc_w1, w1B, nW);
  f32_to_bf16_kernel<<<gW, 256, 0, stream>>>(enc_w2, w2B, nW);
  f32_to_bf16_kernel<<<gW, 256, 0, stream>>>(wq, wqB, nW);
  f32_to_bf16_kernel<<<gW, 256, 0, stream>>>(wk, wkB, nW);
  f32_to_bf16_kernel<<<gW, 256, 0, stream>>>(wv, wvB, nW);
  f32_to_bf16_kernel<<<gW, 256, 0, stream>>>(wo, woB, nW);
  f32_to_bf16_kernel<<<gW, 256, 0, stream>>>(tw2, tw2B, nW);
  const int nW1 = kD * kXDr;
  f32_to_bf16_kernel<<<(nW1 / 4 + 255) / 256, 256, 0, stream>>>(tw1, tw1B, nW1);
  const int nQ1 = 128 * kD;
  f32_to_bf16_kernel<<<(nQ1 / 4 + 255) / 256, 256, 0, stream>>>(qw1, qw1B, nQ1);

  const size_t sh_a = (size_t)16 * kD * 2;   // 16 KB A staging
  kv_proj_kernel<<<kMT, 256, sh_a, stream>>>(wmem, wkB, bk, wvB, bv, kmat, vt);

  const size_t sh_enc = sh_a + 32768 + 16384 + (256 + 256 + 16 + 16) * sizeof(float);
  encoder_kernel<<<kB / 16, 256, sh_enc, stream>>>(obs, w1B, enc_b1, enc_g, enc_be,
                                                   w2B, enc_b2, enc);

  qproj_kernel<<<kB / 16, 256, sh_a, stream>>>(enc, wqB, bq, qmat);

  const size_t sh_attn = 65536 + 32768 + (256 + 16) * sizeof(float);
  attn_kernel<<<dim3(kB / 16, 8), 256, sh_attn, stream>>>(qmat, kmat, vt, ctx);

  const size_t sh_tr = 17408 + 32768 + 16384 + 16384 + 8192 +
                       (256 + 256 + 16 + 16) * sizeof(float);
  transition_kernel<<<kB / 16, 256, sh_tr, stream>>>(ctx, enc, ego, woB, bo,
                                                     tw1B, tb1, tg, tbe, tw2B, tb2,
                                                     qw1B, qb1, qw2, qb2, out_ns, out_q);

  memupd_kernel<<<kM, 128, 0, stream>>>(wmem, out_ns, out_q, uidx, out_mem);
}
